// SkipConnectForwardAggModel_15728170238096
// MI455X (gfx1250) — matrix operand order repaired and checked by reading, not compiled, not run on hardware
//
#include <hip/hip_runtime.h>

// ---------------------------------------------------------------------------
// SkipConnectForwardAggModel for MI455X (gfx1250, wave32, WMMA bf16)
//
// T=128 B=128 IN=256 H=512 L=256, 2-layer LSTM + BN-MLP head.
// All GEMMs run on v_wmma_f32_16x16x32_bf16 with fp32 accumulation.
// LDS staging uses GLOBAL_LOAD_ASYNC_TO_LDS_B128 (ASYNCcnt) when available.
// ---------------------------------------------------------------------------

#define T_ 128
#define B_ 128
#define IN_ 256
#define H_ 512
#define G4_ 2048          // 4*H
#define L_ 256
#define N_ 16384          // T*B
#define BH_ (B_ * H_)

typedef float  v8f   __attribute__((ext_vector_type(8)));
typedef __bf16 v16bf __attribute__((ext_vector_type(16)));
typedef int    v4i_  __attribute__((ext_vector_type(4)));

union FragU { v16bf v; uint4 q[2]; };

// ---------------- async global->LDS staging (CDNA5), with fallback ---------
#if defined(__AMDGCN__) && __has_builtin(__builtin_amdgcn_global_load_async_to_lds_b128)
#define HAVE_ASYNC 1
typedef __attribute__((address_space(1))) v4i_* gptr4;
typedef __attribute__((address_space(3))) v4i_* lptr4;
__device__ __forceinline__ void stage16(const __bf16* g, __bf16* l) {
  __builtin_amdgcn_global_load_async_to_lds_b128((gptr4)g, (lptr4)l, 0, 0);
}
#if __has_builtin(__builtin_amdgcn_s_wait_asynccnt)
#define WAIT_ASYNC(n) __builtin_amdgcn_s_wait_asynccnt(n)
#else
#define WAIT_ASYNC(n) asm volatile("s_wait_asynccnt %0" ::"i"(n) : "memory")
#endif
#else
#define HAVE_ASYNC 0
__device__ __forceinline__ void stage16(const __bf16* g, __bf16* l) {
  *(uint4*)l = *(const uint4*)g;   // sync fallback: global load + ds store
}
#define WAIT_ASYNC(n)
#endif

__device__ __forceinline__ float sigm(float x) {
  return 1.0f / (1.0f + __expf(-x));
}

// ---------------------------------------------------------------------------
// Generic bf16 GEMM: out[M][Nc] = A[M][K] * W[Nc][K]^T + bias[Nc]   (f32 out)
// 256 threads = 8 waves. WG tile 128x128. wave = 16-row stripe x 8 col tiles.
// K chunked by 32, double-buffered LDS (2 x (8KB A + 8KB B)).
// ---------------------------------------------------------------------------
__global__ __launch_bounds__(256) void gemm_bf16_kernel(
    const __bf16* __restrict__ A, const __bf16* __restrict__ W,
    const float* __restrict__ bias, float* __restrict__ out,
    int M, int Nc, int K) {
  __shared__ __bf16 As[2][128 * 32];
  __shared__ __bf16 Bs[2][128 * 32];
  const int tid  = threadIdx.x;
  const int wave = tid >> 5, lane = tid & 31;
  const int mrow = lane & 15, lhi = lane >> 4;
  const int r0 = blockIdx.x * 128;
  const int c0 = blockIdx.y * 128;
  (void)M;

  v8f acc[8];
#pragma unroll
  for (int cs = 0; cs < 8; ++cs) {
    const float bv = bias[c0 + cs * 16 + mrow];
    v8f c;
#pragma unroll
    for (int v = 0; v < 8; ++v) c[v] = bv;
    acc[cs] = c;
  }

  const __bf16* Abase = A + (size_t)r0 * K;
  const __bf16* Wbase = W + (size_t)c0 * K;

  // stage a 128x32 bf16 tile: 512 chunks of 16B, 2 per thread
#define STAGE_TILE(gbase, lds, kb)                                         \
  {                                                                        \
    _Pragma("unroll") for (int i = 0; i < 2; ++i) {                        \
      const int cch = tid + 256 * i;                                       \
      const int r = cch >> 2, part = cch & 3;                              \
      stage16((gbase) + (size_t)r * K + (kb) + part * 8,                   \
              &(lds)[r * 32 + part * 8]);                                  \
    }                                                                      \
  }

  STAGE_TILE(Abase, As[0], 0);
  STAGE_TILE(Wbase, Bs[0], 0);
  const int KC = K >> 5;
  for (int kc = 0; kc < KC; ++kc) {
    const int cur = kc & 1;
    if (kc + 1 < KC) {
      STAGE_TILE(Abase, As[cur ^ 1], (kc + 1) * 32);
      STAGE_TILE(Wbase, Bs[cur ^ 1], (kc + 1) * 32);
      WAIT_ASYNC(4);
    } else {
      WAIT_ASYNC(0);
    }
    __syncthreads();
    FragU fa;
    const __bf16* ab = &As[cur][(wave * 16 + mrow) * 32 + lhi * 8];
    fa.q[0] = *(const uint4*)ab;
    fa.q[1] = *(const uint4*)(ab + 16);
#pragma unroll
    for (int cs = 0; cs < 8; ++cs) {
      FragU fb;
      const __bf16* bb = &Bs[cur][(cs * 16 + mrow) * 32 + lhi * 8];
      fb.q[0] = *(const uint4*)bb;
      fb.q[1] = *(const uint4*)(bb + 16);
      acc[cs] = __builtin_amdgcn_wmma_f32_16x16x32_bf16(
          false, fa.v, false, fb.v, (short)0, acc[cs], false, false);
    }
    __syncthreads();
  }
#undef STAGE_TILE

#pragma unroll
  for (int cs = 0; cs < 8; ++cs)
#pragma unroll
    for (int v = 0; v < 8; ++v) {
      const int row = r0 + wave * 16 + v + 8 * lhi;
      const int col = c0 + cs * 16 + mrow;
      out[(size_t)row * Nc + col] = acc[cs][v];
    }
}

// ---------------------------------------------------------------------------
// One LSTM timestep. Grid = 16 WGs (each owns 32 h-columns -> 128 gate cols).
// gates = xg_t + h_prev @ W_hh^T (WMMA, K=512). Cell update fused in-register:
// tile col-stripe cs -> gate group g=cs/2, so i/f/g/o for a (row,hcol) sit in
// the same lane/VGPR slot of acc[0/2/4/6 + j].
// ---------------------------------------------------------------------------
__global__ __launch_bounds__(256) void lstm_step_kernel(
    const float* __restrict__ xg_t,   // [B][4H] f32 (includes both biases)
    const __bf16* __restrict__ Whh,   // [4H][H] bf16
    const __bf16* __restrict__ hin,   // [B][H] bf16  (h_{t-1})
    __bf16* __restrict__ hout_bf,     // [B][H] bf16  (h_t, other buffer)
    float* __restrict__ c_state,      // [B][H] f32
    const float* __restrict__ h1_t,   // layer1: h1s_t (for skip-sum); else 0
    float* __restrict__ fout,         // layer0: h1s_t ; layer1: agg_t (d_out)
    __bf16* __restrict__ fbf) {       // layer0: h1s_bf_t ; layer1: agg_bf_t
  __shared__ __bf16 As[2][128 * 32];
  __shared__ __bf16 Bs[2][128 * 32];
  const int tid  = threadIdx.x;
  const int wave = tid >> 5, lane = tid & 31;
  const int mrow = lane & 15, lhi = lane >> 4;
  const int hbase = blockIdx.x * 32;

  // init accumulators from xg_t (biases already folded in)
  v8f acc[8];
#pragma unroll
  for (int cs = 0; cs < 8; ++cs) {
    const int gc = (cs >> 1) * H_ + hbase + (cs & 1) * 16 + mrow;
    v8f c;
#pragma unroll
    for (int v = 0; v < 8; ++v) {
      const int row = wave * 16 + v + 8 * lhi;
      c[v] = xg_t[(size_t)row * G4_ + gc];
    }
    acc[cs] = c;
  }

#define STAGE_A(kb, buf)                                                   \
  {                                                                        \
    _Pragma("unroll") for (int i = 0; i < 2; ++i) {                        \
      const int cch = tid + 256 * i;                                       \
      const int r = cch >> 2, part = cch & 3;                              \
      stage16(hin + (size_t)r * H_ + (kb) + part * 8,                      \
              &As[buf][r * 32 + part * 8]);                                \
    }                                                                      \
  }
#define STAGE_B(kb, buf)                                                   \
  {                                                                        \
    _Pragma("unroll") for (int i = 0; i < 2; ++i) {                        \
      const int cch = tid + 256 * i;                                       \
      const int q = cch >> 2, part = cch & 3;                              \
      const int wr = (q >> 5) * H_ + hbase + (q & 31);                     \
      stage16(Whh + (size_t)wr * H_ + (kb) + part * 8,                     \
              &Bs[buf][q * 32 + part * 8]);                                \
    }                                                                      \
  }

  STAGE_A(0, 0);
  STAGE_B(0, 0);
  const int KC = H_ / 32;  // 16
  for (int kc = 0; kc < KC; ++kc) {
    const int cur = kc & 1;
    if (kc + 1 < KC) {
      STAGE_A((kc + 1) * 32, cur ^ 1);
      STAGE_B((kc + 1) * 32, cur ^ 1);
      WAIT_ASYNC(4);
    } else {
      WAIT_ASYNC(0);
    }
    __syncthreads();
    FragU fa;
    const __bf16* ab = &As[cur][(wave * 16 + mrow) * 32 + lhi * 8];
    fa.q[0] = *(const uint4*)ab;
    fa.q[1] = *(const uint4*)(ab + 16);
#pragma unroll
    for (int cs = 0; cs < 8; ++cs) {
      FragU fb;
      const __bf16* bb = &Bs[cur][(cs * 16 + mrow) * 32 + lhi * 8];
      fb.q[0] = *(const uint4*)bb;
      fb.q[1] = *(const uint4*)(bb + 16);
      acc[cs] = __builtin_amdgcn_wmma_f32_16x16x32_bf16(
          false, fa.v, false, fb.v, (short)0, acc[cs], false, false);
    }
    __syncthreads();
  }
#undef STAGE_A
#undef STAGE_B

  // fused LSTM cell update (i,f,g,o live in acc[0/2/4/6 + j])
#pragma unroll
  for (int j = 0; j < 2; ++j) {
    const int hc = hbase + j * 16 + mrow;
    v8f gi = acc[0 + j], gf = acc[2 + j], gg = acc[4 + j], go = acc[6 + j];
#pragma unroll
    for (int v = 0; v < 8; ++v) {
      const int row = wave * 16 + v + 8 * lhi;
      const size_t idx = (size_t)row * H_ + hc;
      const float cprev = c_state[idx];
      const float iv = sigm(gi[v]);
      const float fv = sigm(gf[v]);
      const float gv = tanhf(gg[v]);
      const float ov = sigm(go[v]);
      const float cn = fv * cprev + iv * gv;
      const float hn = ov * tanhf(cn);
      c_state[idx] = cn;
      hout_bf[idx] = (__bf16)hn;
      const float o = h1_t ? (h1_t[idx] + hn) : hn;
      fout[idx] = o;
      fbf[idx]  = (__bf16)o;
    }
  }
}

// ------------------------- small helper kernels ----------------------------
__global__ void cvt_f32_bf16_kernel(const float* __restrict__ in,
                                    __bf16* __restrict__ out, int n4) {
  const int i = blockIdx.x * blockDim.x + threadIdx.x;
  if (i < n4) {
    const float4 v = ((const float4*)in)[i];
    out[4 * i + 0] = (__bf16)v.x;
    out[4 * i + 1] = (__bf16)v.y;
    out[4 * i + 2] = (__bf16)v.z;
    out[4 * i + 3] = (__bf16)v.w;
  }
}

__global__ void bias_sum_kernel(const float* __restrict__ a,
                                const float* __restrict__ b,
                                float* __restrict__ out, int n) {
  const int i = blockIdx.x * blockDim.x + threadIdx.x;
  if (i < n) out[i] = a[i] + b[i];
}

__global__ void zero_state_kernel(float* __restrict__ c,
                                  __bf16* __restrict__ hbuf) {
  const int i = blockIdx.x * blockDim.x + threadIdx.x;  // BH threads
  c[i] = 0.0f;
  hbuf[i] = (__bf16)0.0f;
  hbuf[i + BH_] = (__bf16)0.0f;
}

// BN training stats per column over N rows; fold gamma/beta into (a,b):
// y = a[col]*z + b[col]. Grid = H/64 blocks, 256 threads (64 cols x 4 rows).
__global__ __launch_bounds__(256) void bn_stats_kernel(
    const float* __restrict__ z1, const float* __restrict__ gamma,
    const float* __restrict__ beta, float* __restrict__ ab) {
  const int c0 = blockIdx.x * 64;
  const int col = threadIdx.x & 63;
  const int rp = threadIdx.x >> 6;
  float s = 0.f, s2 = 0.f;
  for (int r = rp; r < N_; r += 4) {
    const float v = z1[(size_t)r * H_ + c0 + col];
    s += v;
    s2 += v * v;
  }
  __shared__ float sh[2][4][64];
  sh[0][rp][col] = s;
  sh[1][rp][col] = s2;
  __syncthreads();
  if (threadIdx.x < 64) {
    float ts = 0.f, ts2 = 0.f;
#pragma unroll
    for (int i = 0; i < 4; ++i) {
      ts += sh[0][i][threadIdx.x];
      ts2 += sh[1][i][threadIdx.x];
    }
    const float mu = ts * (1.0f / N_);
    const float var = ts2 * (1.0f / N_) - mu * mu;
    const float rs = rsqrtf(var + 1e-5f);
    const float a = gamma[c0 + threadIdx.x] * rs;
    ab[c0 + threadIdx.x] = a;
    ab[H_ + c0 + threadIdx.x] = beta[c0 + threadIdx.x] - a * mu;
  }
}

__global__ void bn_apply_relu_kernel(const float* __restrict__ z1,
                                     const float* __restrict__ ab,
                                     __bf16* __restrict__ zr) {
  const size_t i = (size_t)blockIdx.x * blockDim.x + threadIdx.x;  // N*H
  const int col = (int)(i & (H_ - 1));
  const float v = ab[col] * z1[i] + ab[H_ + col];
  zr[i] = (__bf16)fmaxf(v, 0.0f);
}

// ------------------------------ host side ----------------------------------
static inline void launch_cvt(const float* in, __bf16* out, int n,
                              hipStream_t s) {
  const int n4 = n / 4;
  cvt_f32_bf16_kernel<<<(n4 + 255) / 256, 256, 0, s>>>(in, out, n4);
}

extern "C" void kernel_launch(void* const* d_in, const int* in_sizes, int n_in,
                              void* d_out, int out_size, void* d_ws,
                              size_t ws_size, hipStream_t stream) {
  (void)in_sizes; (void)n_in; (void)out_size; (void)ws_size;
  const float* x     = (const float*)d_in[0];
  const float* W_ih0 = (const float*)d_in[1];
  const float* W_hh0 = (const float*)d_in[2];
  const float* b_ih0 = (const float*)d_in[3];
  const float* b_hh0 = (const float*)d_in[4];
  const float* W_ih1 = (const float*)d_in[5];
  const float* W_hh1 = (const float*)d_in[6];
  const float* b_ih1 = (const float*)d_in[7];
  const float* b_hh1 = (const float*)d_in[8];
  const float* W1    = (const float*)d_in[9];
  const float* b1    = (const float*)d_in[10];
  const float* gamma = (const float*)d_in[11];
  const float* beta  = (const float*)d_in[12];
  const float* W2    = (const float*)d_in[13];
  const float* b2    = (const float*)d_in[14];
  float* out = (float*)d_out;  // [N*L] preds then [N*H] agg

  char* ws = (char*)d_ws;      // ~256.3 MB used
  float*  xg      = (float*)(ws + 0);                 // [N][4H]  f32 128MB
  float*  h1s     = (float*)(ws + 134217728);         // [N][H]   f32  32MB
  __bf16* h1s_bf  = (__bf16*)(ws + 167772160);        // [N][H]   bf16 16MB
  __bf16* x_bf    = (__bf16*)(ws + 184549376);        // [N][IN]  bf16  8MB
  __bf16* agg_bf  = (__bf16*)(ws + 192937984);        // [N][H]   bf16 16MB
  __bf16* zr_bf   = (__bf16*)(ws + 209715200);        // [N][H]   bf16 16MB
  float*  z1      = (float*)(ws + 226492416);         // [N][H]   f32  32MB
  __bf16* Wih0_bf = (__bf16*)(ws + 260046848);        // [4H][IN]
  __bf16* Whh0_bf = (__bf16*)(ws + 261095424);        // [4H][H]
  __bf16* Wih1_bf = (__bf16*)(ws + 263192576);        // [4H][H]
  __bf16* Whh1_bf = (__bf16*)(ws + 265289728);        // [4H][H]
  __bf16* W1_bf   = (__bf16*)(ws + 267386880);        // [H][H]
  __bf16* W2_bf   = (__bf16*)(ws + 267911168);        // [L][H]
  float*  bias0   = (float*)(ws + 268173312);         // [4H]
  float*  bias1   = (float*)(ws + 268181504);         // [4H]
  float*  ab      = (float*)(ws + 268189696);         // [2][H]
  float*  c_state = (float*)(ws + 268193792);         // [B][H] f32
  __bf16* hbuf    = (__bf16*)(ws + 268455936);        // [2][B][H] bf16

  // --- convert weights/inputs to bf16 once per call ---
  launch_cvt(x,     x_bf,    N_ * IN_,   stream);
  launch_cvt(W_ih0, Wih0_bf, G4_ * IN_,  stream);
  launch_cvt(W_hh0, Whh0_bf, G4_ * H_,   stream);
  launch_cvt(W_ih1, Wih1_bf, G4_ * H_,   stream);
  launch_cvt(W_hh1, Whh1_bf, G4_ * H_,   stream);
  launch_cvt(W1,    W1_bf,   H_ * H_,    stream);
  launch_cvt(W2,    W2_bf,   L_ * H_,    stream);
  bias_sum_kernel<<<G4_ / 256, 256, 0, stream>>>(b_ih0, b_hh0, bias0, G4_);
  bias_sum_kernel<<<G4_ / 256, 256, 0, stream>>>(b_ih1, b_hh1, bias1, G4_);

  // --- layer 0: xg = x @ W_ih0^T + b_ih0 + b_hh0 ---
  gemm_bf16_kernel<<<dim3(N_ / 128, G4_ / 128), 256, 0, stream>>>(
      x_bf, Wih0_bf, bias0, xg, N_, G4_, IN_);
  zero_state_kernel<<<BH_ / 256, 256, 0, stream>>>(c_state, hbuf);
  for (int t = 0; t < T_; ++t) {
    lstm_step_kernel<<<H_ / 32, 256, 0, stream>>>(
        xg + (size_t)t * B_ * G4_, Whh0_bf, hbuf + (t & 1) * BH_,
        hbuf + ((t + 1) & 1) * BH_, c_state, nullptr,
        h1s + (size_t)t * BH_, h1s_bf + (size_t)t * BH_);
  }

  // --- layer 1: xg = h1s @ W_ih1^T + b_ih1 + b_hh1 (xg buffer reused) ---
  gemm_bf16_kernel<<<dim3(N_ / 128, G4_ / 128), 256, 0, stream>>>(
      h1s_bf, Wih1_bf, bias1, xg, N_, G4_, H_);
  zero_state_kernel<<<BH_ / 256, 256, 0, stream>>>(c_state, hbuf);
  float* agg_out = out + (size_t)N_ * L_;  // second output: agg [T,B,H]
  for (int t = 0; t < T_; ++t) {
    lstm_step_kernel<<<H_ / 32, 256, 0, stream>>>(
        xg + (size_t)t * B_ * G4_, Whh1_bf, hbuf + (t & 1) * BH_,
        hbuf + ((t + 1) & 1) * BH_, c_state, h1s + (size_t)t * BH_,
        agg_out + (size_t)t * BH_, agg_bf + (size_t)t * BH_);
  }

  // --- head: Linear -> BN(train) -> ReLU -> Linear ---
  gemm_bf16_kernel<<<dim3(N_ / 128, H_ / 128), 256, 0, stream>>>(
      agg_bf, W1_bf, b1, z1, N_, H_, H_);
  bn_stats_kernel<<<H_ / 64, 256, 0, stream>>>(z1, gamma, beta, ab);
  bn_apply_relu_kernel<<<(N_ * H_) / 256, 256, 0, stream>>>(z1, ab, zr_bf);
  gemm_bf16_kernel<<<dim3(N_ / 128, L_ / 128), 256, 0, stream>>>(
      zr_bf, W2_bf, b2, out, N_, L_, H_);
}
